// BatchedCompressedLinear_26517128085921
// MI455X (gfx1250) — compile-verified, matrix operand order checked by reading, not run on hardware
//
#include <hip/hip_runtime.h>

// ---------------------------------------------------------------------------
// y[M,N] = x[M,K] @ W^T + bias,  W = w_q(int8 codes as int32) * scale
// M = B*S = 8192, K = IN = 4096, N = OUT = 16384
//
// Memory-bound (~940 MB HBM vs ~1.1 TFLOP -> ~40us floor @ 23.3 TB/s).
//  - v_wmma_f32_16x16x32_bf16; int8 weight codes are EXACT in bf16
//  - split-bf16 (hi+lo) activations -> ~fp32 accuracy at 2x (free) compute
//  - BK=64: 2 WMMA K-steps per LDS fill, half the barriers of BK=32
//  - global loads staged in registers BEFORE the barrier so their latency
//    overlaps the previous iteration's WMMA tail
//  - M-tiles on blockIdx.x: consecutive WGs share the W N-tile (L2 reuse),
//    x (134MB) stays resident in the 192MB L2
// ---------------------------------------------------------------------------

typedef __attribute__((ext_vector_type(16))) __bf16 v16bf;
typedef __attribute__((ext_vector_type(8)))  __bf16 v8bf;
typedef __attribute__((ext_vector_type(4)))  __bf16 v4bf;
typedef __attribute__((ext_vector_type(8)))  float  v8f;

#define M_DIM 8192
#define N_DIM 16384
#define K_DIM 4096

#define BM 128
#define BN 128
#define BK 64
#define LDSK 72   // padded LDS row pitch (bf16): 144B rows, 16B-aligned, bank-spread

__global__ __launch_bounds__(256)
void BatchedCompressedLinear_wmma_kernel(const float* __restrict__ X,
                                         const int*   __restrict__ Wq,
                                         const float* __restrict__ scale,
                                         const float* __restrict__ bias,
                                         float*       __restrict__ Y)
{
    __shared__ __bf16 lXhi[BM * LDSK];
    __shared__ __bf16 lXlo[BM * LDSK];
    __shared__ __bf16 lW  [BN * LDSK];

    const int tid   = threadIdx.x;
    const int lane  = tid & 31;
    const int wave  = tid >> 5;
    const int waveM = wave & 1;   // 2 wave-rows  (64 rows each)
    const int waveN = wave >> 1;  // 4 wave-cols  (32 cols each)
    const int half  = lane >> 4;  // WMMA K-interleave lane group
    const int l16   = lane & 15;

    const int tileM = blockIdx.x * BM;   // fast-varying -> W tile reused in L2
    const int tileN = blockIdx.y * BN;

    // per-thread cooperative-load geometry: 128 rows x 64 K = 2048 float4
    // chunks, 8 per thread
    const int c0   = tid;
    v8f acc[4][2];
#pragma unroll
    for (int mt = 0; mt < 4; ++mt)
#pragma unroll
        for (int nt = 0; nt < 2; ++nt)
            acc[mt][nt] = (v8f)(0.0f);

    for (int kk = 0; kk < K_DIM; kk += BK) {
        // ---- stage global loads in registers (before any barrier) ---------
        float4 xr[8];
        int4   wr[8];
#pragma unroll
        for (int i = 0; i < 8; ++i) {
            const int c   = c0 + i * 256;
            const int r   = c >> 4;
            const int col = (c & 15) << 2;
            const float* xp = X  + (size_t)(tileM + r) * K_DIM + kk + col;
            const int*   wp = Wq + (size_t)(tileN + r) * K_DIM + kk + col;
            xr[i] = *(const float4*)xp;
            wr[i] = *(const int4*)wp;
            // speculative prefetch of next K-slab (gfx1250 global_prefetch_b8;
            // OOB translation failures are silently dropped)
            __builtin_prefetch(xp + BK, 0, 0);
            __builtin_prefetch(wp + BK, 0, 0);
        }

        __syncthreads();   // previous iteration's fragment readers are done

        // ---- convert + store to LDS: x -> bf16 hi/lo, w codes -> bf16 -----
#pragma unroll
        for (int i = 0; i < 8; ++i) {
            const int c   = c0 + i * 256;
            const int r   = c >> 4;
            const int col = (c & 15) << 2;

            const float4 xv = xr[i];
            __bf16 h0 = (__bf16)xv.x, h1 = (__bf16)xv.y,
                   h2 = (__bf16)xv.z, h3 = (__bf16)xv.w;
            v4bf hi = { h0, h1, h2, h3 };
            v4bf lo = { (__bf16)(xv.x - (float)h0), (__bf16)(xv.y - (float)h1),
                        (__bf16)(xv.z - (float)h2), (__bf16)(xv.w - (float)h3) };
            *(v4bf*)(lXhi + r * LDSK + col) = hi;
            *(v4bf*)(lXlo + r * LDSK + col) = lo;

            const int4 wv = wr[i];
            v4bf wb = { (__bf16)(float)wv.x, (__bf16)(float)wv.y,
                        (__bf16)(float)wv.z, (__bf16)(float)wv.w };
            *(v4bf*)(lW + r * LDSK + col) = wb;
        }

        __syncthreads();

        // ---- 2 WMMA K-steps per LDS fill ----------------------------------
#pragma unroll
        for (int ks = 0; ks < BK; ks += 32) {
            // B fragments (N side). CDNA5 bf16 layout: lanes 0-15 hold
            // K 0-7 then 16-23; lanes 16-31 hold K 8-15 then 24-31.
            v16bf bfrag[2];
#pragma unroll
            for (int nt = 0; nt < 2; ++nt) {
                const __bf16* rp =
                    lW + (waveN * 32 + nt * 16 + l16) * LDSK + ks + half * 8;
                union { v16bf v; v8bf h[2]; } u;
                u.h[0] = *(const v8bf*)(rp);
                u.h[1] = *(const v8bf*)(rp + 16);
                bfrag[nt] = u.v;
            }

#pragma unroll
            for (int mt = 0; mt < 4; ++mt) {
                const int arow =
                    (waveM * 64 + mt * 16 + l16) * LDSK + ks + half * 8;

                union { v16bf v; v8bf h[2]; } ua;
                ua.h[0] = *(const v8bf*)(lXhi + arow);
                ua.h[1] = *(const v8bf*)(lXhi + arow + 16);
                const v16bf ahi = ua.v;

                union { v16bf v; v8bf h[2]; } ub;
                ub.h[0] = *(const v8bf*)(lXlo + arow);
                ub.h[1] = *(const v8bf*)(lXlo + arow + 16);
                const v16bf alo = ub.v;

#pragma unroll
                for (int nt = 0; nt < 2; ++nt) {
                    acc[mt][nt] = __builtin_amdgcn_wmma_f32_16x16x32_bf16(
                        false, ahi, false, bfrag[nt], (short)0, acc[mt][nt],
                        false, false);
                    acc[mt][nt] = __builtin_amdgcn_wmma_f32_16x16x32_bf16(
                        false, alo, false, bfrag[nt], (short)0, acc[mt][nt],
                        false, false);
                }
            }
        }
    }

    // ---- epilogue: y = acc * scale + bias ---------------------------------
    const float s = scale[0];
#pragma unroll
    for (int nt = 0; nt < 2; ++nt) {
        const int gcol = tileN + waveN * 32 + nt * 16 + l16;
        const float bv = bias[gcol];
#pragma unroll
        for (int mt = 0; mt < 4; ++mt) {
#pragma unroll
            for (int i = 0; i < 8; ++i) {
                // C layout: VGPR i -> M = i (+8 for lanes 16-31), N = lane%16
                const int grow = tileM + waveM * 64 + mt * 16 + i + half * 8;
                Y[(size_t)grow * N_DIM + gcol] = acc[mt][nt][i] * s + bv;
            }
        }
    }
}

extern "C" void kernel_launch(void* const* d_in, const int* in_sizes, int n_in,
                              void* d_out, int out_size, void* d_ws, size_t ws_size,
                              hipStream_t stream) {
    const float* x     = (const float*)d_in[0];
    const int*   w_q   = (const int*)  d_in[1];
    const float* scale = (const float*)d_in[2];
    const float* bias  = (const float*)d_in[3];
    float*       y     = (float*)d_out;

    dim3 grid(M_DIM / BM, N_DIM / BN);   // (64, 128): M fast -> W reuse in L2
    dim3 block(256);                     // 8 wave32 waves per WGP workgroup
    BatchedCompressedLinear_wmma_kernel<<<grid, block, 0, stream>>>(x, w_q, scale, bias, y);
}
